// AttackHead_9302899163211
// MI455X (gfx1250) — compile-verified, matrix operand order checked by reading, not run on hardware
//
#include <hip/hip_runtime.h>
#include <hip/hip_bf16.h>

#define EDIM 256
#define BDIM 16
#define SDIM 2048
#define NSELFD 8
#define ADIM 3
#define NEGV (-1000000000.0f)

typedef __attribute__((ext_vector_type(16))) _Float16 v16h;
typedef __attribute__((ext_vector_type(8)))  _Float16 v8h;
typedef __attribute__((ext_vector_type(8)))  float    v8f;

union Frag16 { v16h v; v8h h2[2]; };

// Async DMA global->LDS, 16 bytes per lane (ASYNCcnt-tracked, ISA 10.x/15.18).
#define ASYNC_LDS_LOAD_B128(ldsoff, gaddr)                                \
  asm volatile("global_load_async_to_lds_b128 %0, %1, off"                \
               :: "v"(ldsoff), "v"(gaddr) : "memory")

// ---------------------------------------------------------------------------
// Generic K=256 GEMM (synchronous staging, f32 or f16 A with on-the-fly f16
// conversion): Out[M x 256] = act(A @ W + bias). Block 128(M) x 64(N), 8
// waves, each wave a 16x64 strip via 4x v_wmma_f32_16x16x32_f16 per K-step.
// Fragment layouts per ISA 7.12.2 (each half = 8 contiguous f16 = 1x b128).
// ---------------------------------------------------------------------------
template<bool A_F16, bool RELU, bool OUT_F16>
__global__ __launch_bounds__(256) void gemm256_wmma(
    const void* __restrict__ Aptr, const float* __restrict__ W,
    const float* __restrict__ bias, void* __restrict__ Outptr, int M)
{
  __shared__ __align__(16) _Float16 sA[128][72];
  __shared__ __align__(16) _Float16 sW[64][72];

  const int tid  = threadIdx.x;
  const int m0   = blockIdx.x * 128;
  const int n0   = blockIdx.y * 64;
  const int wave = tid >> 5;
  const int lane = tid & 31;
  const int mw   = wave * 16;
  const int lr   = lane & 15;
  const int khalf = (lane >> 4) * 8;

  const float*    Af = (const float*)Aptr;
  const _Float16* Ah = (const _Float16*)Aptr;

  v8f acc[4] = {};

  for (int k0 = 0; k0 < EDIM; k0 += 64) {
    #pragma unroll
    for (int i = 0; i < 32; ++i) {
      int idx = tid + i * 256;
      int r = idx >> 6, c = idx & 63;
      int gm = m0 + r;
      float v = 0.f;
      if (gm < M) v = A_F16 ? (float)Ah[(size_t)gm * EDIM + k0 + c]
                            : Af[(size_t)gm * EDIM + k0 + c];
      sA[r][c] = (_Float16)v;
    }
    #pragma unroll
    for (int i = 0; i < 16; ++i) {
      int idx = tid + i * 256;
      int n = idx & 63, kk = idx >> 6;
      sW[n][kk] = (_Float16)W[(size_t)(k0 + kk) * EDIM + n0 + n];
    }
    __syncthreads();

    #pragma unroll
    for (int kk = 0; kk < 64; kk += 32) {
      Frag16 a;
      a.h2[0] = *(const v8h*)&sA[mw + lr][kk + khalf];
      a.h2[1] = *(const v8h*)&sA[mw + lr][kk + 16 + khalf];
      #pragma unroll
      for (int nt = 0; nt < 4; ++nt) {
        Frag16 bfr;
        bfr.h2[0] = *(const v8h*)&sW[nt * 16 + lr][kk + khalf];
        bfr.h2[1] = *(const v8h*)&sW[nt * 16 + lr][kk + 16 + khalf];
        acc[nt] = __builtin_amdgcn_wmma_f32_16x16x32_f16(
            false, a.v, false, bfr.v, (short)0, acc[nt], false, false);
      }
    }
    __syncthreads();
  }

  const int mrow = mw + (lane >> 4) * 8;
  #pragma unroll
  for (int nt = 0; nt < 4; ++nt) {
    int n = n0 + nt * 16 + lr;
    float bv = bias ? bias[n] : 0.f;
    #pragma unroll
    for (int i = 0; i < 8; ++i) {
      int gm = m0 + mrow + i;
      if (gm < M) {
        float v = acc[nt][i] + bv;
        if (RELU) v = v > 0.f ? v : 0.f;
        if (OUT_F16) ((_Float16*)Outptr)[(size_t)gm * EDIM + n] = (_Float16)v;
        else         ((float*)Outptr)[(size_t)gm * EDIM + n] = v;
      }
    }
  }
}

// Wt[n][k] = (f16) W[k*256 + n]  (one 256x256 slab, coalesced reads)
__global__ void transpose_w_f16(const float* __restrict__ W,
                                _Float16* __restrict__ Wt)
{
  int idx = blockIdx.x * 256 + threadIdx.x;    // 65536 total
  int n = idx & 255, k = idx >> 8;
  Wt[(size_t)n * EDIM + k] = (_Float16)W[(size_t)k * EDIM + n];
}

// ---------------------------------------------------------------------------
// Stage-2 main GEMM: Out_f16[M x 256] = A_f16 @ Wt^T.  A and Wt tiles are
// DMA'd into double-buffered LDS with global_load_async_to_lds_b128 and
// drained with s_wait_asynccnt (in-order completion: waiting "<=6" leaves
// exactly the 6 just-issued next-chunk ops in flight, overlapping the WMMAs).
// Requires M % 128 == 0 (used only for M = 32768).
// ---------------------------------------------------------------------------
__global__ __launch_bounds__(256) void gemm256_wmma_async(
    const _Float16* __restrict__ A,     // M x 256, row-major
    const _Float16* __restrict__ Wt,    // 256(n) x 256(k), pre-transposed f16
    _Float16* __restrict__ Out, int M)
{
  __shared__ __align__(16) _Float16 sA[2][128][72];   // 36.0 KB
  __shared__ __align__(16) _Float16 sW[2][64][72];    // 18.0 KB

  const int tid  = threadIdx.x;
  const int m0   = blockIdx.x * 128;
  const int n0   = blockIdx.y * 64;
  const int wave = tid >> 5;
  const int lane = tid & 31;
  const int mw   = wave * 16;
  const int lr   = lane & 15;
  const int khalf = (lane >> 4) * 8;
  (void)M;

  // 6 async b128 ops per thread per K-chunk: 4 for A (128x64 f16), 2 for W.
  auto issue_chunk = [&](int kt, int bsel) {
    const int k0 = kt * 64;
    #pragma unroll
    for (int i = 0; i < 4; ++i) {
      int id = tid + i * 256;
      int r = id >> 3, c8 = (id & 7) * 8;
      unsigned lo = (unsigned)(size_t)&sA[bsel][r][c8];
      unsigned long long ga =
          (unsigned long long)(size_t)(A + (size_t)(m0 + r) * EDIM + k0 + c8);
      ASYNC_LDS_LOAD_B128(lo, ga);
    }
    #pragma unroll
    for (int i = 0; i < 2; ++i) {
      int id = tid + i * 256;
      int r = id >> 3, c8 = (id & 7) * 8;
      unsigned lo = (unsigned)(size_t)&sW[bsel][r][c8];
      unsigned long long ga =
          (unsigned long long)(size_t)(Wt + (size_t)(n0 + r) * EDIM + k0 + c8);
      ASYNC_LDS_LOAD_B128(lo, ga);
    }
  };

  v8f acc[4] = {};
  issue_chunk(0, 0);

  for (int kt = 0; kt < 4; ++kt) {
    const int cur = kt & 1;
    if (kt < 3) {
      issue_chunk(kt + 1, cur ^ 1);
      asm volatile("s_wait_asynccnt 0x6" ::: "memory");  // chunk kt resident
    } else {
      asm volatile("s_wait_asynccnt 0x0" ::: "memory");
    }
    __syncthreads();

    #pragma unroll
    for (int kk = 0; kk < 64; kk += 32) {
      Frag16 a;
      a.h2[0] = *(const v8h*)&sA[cur][mw + lr][kk + khalf];
      a.h2[1] = *(const v8h*)&sA[cur][mw + lr][kk + 16 + khalf];
      #pragma unroll
      for (int nt = 0; nt < 4; ++nt) {
        Frag16 bfr;
        bfr.h2[0] = *(const v8h*)&sW[cur][nt * 16 + lr][kk + khalf];
        bfr.h2[1] = *(const v8h*)&sW[cur][nt * 16 + lr][kk + 16 + khalf];
        acc[nt] = __builtin_amdgcn_wmma_f32_16x16x32_f16(
            false, a.v, false, bfr.v, (short)0, acc[nt], false, false);
      }
    }
    __syncthreads();   // no wave may DMA into buf[cur] before all finished it
  }

  const int mrow = mw + (lane >> 4) * 8;
  #pragma unroll
  for (int nt = 0; nt < 4; ++nt) {
    int n = n0 + nt * 16 + lr;
    #pragma unroll
    for (int i = 0; i < 8; ++i) {
      int gm = m0 + mrow + i;
      Out[(size_t)gm * EDIM + n] = (_Float16)acc[nt][i];
    }
  }
}

// base[b][n][e] = L[b][e] + SE[b][n][e] + b1[e]
__global__ void prep_base(const float* __restrict__ L, const float* __restrict__ SE,
                          const float* __restrict__ b1, float* __restrict__ base)
{
  int idx = blockIdx.x * 256 + threadIdx.x;
  int e = idx & 255;
  int bn = idx >> 8;
  int b = bn >> 3;
  base[idx] = L[b * 256 + e] + SE[bn * 256 + e] + b1[e];
}

// ---------------------------------------------------------------------------
// Fused head, one block per (b,n): H = relu(EN_f16[b,s,:] + base) built
// straight into A-fragments, WMMA vs zero-padded W2^T, masked logits in LDS,
// block softmax + 3-way argmax. EN rows for the next s-tile are prefetched.
// ---------------------------------------------------------------------------
__global__ __launch_bounds__(256) void attack_fused(
    const _Float16* __restrict__ EN, const float* __restrict__ base,
    const float* __restrict__ W2, const float* __restrict__ b2,
    const unsigned char* __restrict__ avail,
    int* __restrict__ act_out, float* __restrict__ prob_out)
{
  __shared__ __align__(16) _Float16 sW2[16][264];
  __shared__ float sBase[EDIM];
  __shared__ float sLog[SDIM * ADIM];
  __shared__ float sRed[256];

  const int tid = threadIdx.x;
  const int bn  = blockIdx.x;
  const int b   = bn >> 3;

  sBase[tid] = base[bn * EDIM + tid];
  #pragma unroll
  for (int i = 0; i < 16; ++i) {
    int idx = tid + i * 256;
    int nn = idx >> 8, k = idx & 255;
    sW2[nn][k] = (nn < ADIM) ? (_Float16)W2[k * ADIM + nn] : (_Float16)0.f;
  }
  __syncthreads();

  const int wave = tid >> 5, lane = tid & 31;
  const int lr = lane & 15;
  const int khalf = (lane >> 4) * 8;

  for (int it = 0; it < SDIM / (16 * 8); ++it) {
    const int s0 = (it * 8 + wave) * 16;
    const _Float16* enrow = EN + ((size_t)(b * SDIM + s0 + lr)) * EDIM;
    if (it < 15) __builtin_prefetch(enrow + (size_t)128 * EDIM, 0, 0);
    v8f acc = {};
    #pragma unroll
    for (int k0 = 0; k0 < EDIM; k0 += 32) {
      Frag16 a, bfr;
      {
        v8h x = *(const v8h*)(enrow + k0 + khalf);
        #pragma unroll
        for (int j = 0; j < 8; ++j) {
          float v = (float)x[j] + sBase[k0 + khalf + j];
          a.h2[0][j] = (_Float16)(v > 0.f ? v : 0.f);
        }
      }
      {
        v8h x = *(const v8h*)(enrow + k0 + 16 + khalf);
        #pragma unroll
        for (int j = 0; j < 8; ++j) {
          float v = (float)x[j] + sBase[k0 + 16 + khalf + j];
          a.h2[1][j] = (_Float16)(v > 0.f ? v : 0.f);
        }
      }
      bfr.h2[0] = *(const v8h*)&sW2[lr][k0 + khalf];
      bfr.h2[1] = *(const v8h*)&sW2[lr][k0 + 16 + khalf];
      acc = __builtin_amdgcn_wmma_f32_16x16x32_f16(
          false, a.v, false, bfr.v, (short)0, acc, false, false);
    }
    if (lr < ADIM) {
      float bv = b2[lr];
      #pragma unroll
      for (int i = 0; i < 8; ++i) {
        int s = s0 + (lane >> 4) * 8 + i;
        float v = acc[i] + bv;
        bool av = avail[(size_t)bn * SDIM + s] != 0;
        sLog[s * ADIM + lr] = av ? v : NEGV;
      }
    }
  }
  __syncthreads();

  float lmax = -3.4e38f;
  for (int i = tid; i < SDIM * ADIM; i += 256) lmax = fmaxf(lmax, sLog[i]);
  sRed[tid] = lmax; __syncthreads();
  for (int s = 128; s > 0; s >>= 1) {
    if (tid < s) sRed[tid] = fmaxf(sRed[tid], sRed[tid + s]);
    __syncthreads();
  }
  const float gmax = sRed[0]; __syncthreads();

  float lsum = 0.f;
  for (int i = tid; i < SDIM * ADIM; i += 256) lsum += __expf(sLog[i] - gmax);
  sRed[tid] = lsum; __syncthreads();
  for (int s = 128; s > 0; s >>= 1) {
    if (tid < s) sRed[tid] += sRed[tid + s];
    __syncthreads();
  }
  const float inv = 1.f / sRed[0];

  float* pout = prob_out + (size_t)bn * (SDIM * ADIM);
  for (int i = tid; i < SDIM * ADIM; i += 256)
    pout[i] = __expf(sLog[i] - gmax) * inv;

  int* aout = act_out + bn * SDIM;
  for (int s = tid; s < SDIM; s += 256) {
    float v0 = sLog[s * 3], v1 = sLog[s * 3 + 1], v2 = sLog[s * 3 + 2];
    int am = 0; float vm = v0;
    if (v1 > vm) { vm = v1; am = 1; }
    if (v2 > vm) { vm = v2; am = 2; }
    aout[s] = am;
  }
}

extern "C" void kernel_launch(void* const* d_in, const int* in_sizes, int n_in,
                              void* d_out, int out_size, void* d_ws, size_t ws_size,
                              hipStream_t stream) {
  const float* lstm   = (const float*)d_in[0];
  const float* entity = (const float*)d_in[1];
  const float* selfe  = (const float*)d_in[2];
  const unsigned char* avail = (const unsigned char*)d_in[3];
  const float* W_lstm = (const float*)d_in[4];
  const float* b_lstm = (const float*)d_in[5];
  const float* W_en   = (const float*)d_in[6];
  const float* b_en   = (const float*)d_in[7];
  const float* W_self = (const float*)d_in[8];
  const float* b_self = (const float*)d_in[9];
  const float* W1     = (const float*)d_in[10];
  const float* b1     = (const float*)d_in[11];
  const float* W2     = (const float*)d_in[12];
  const float* b2     = (const float*)d_in[13];
  (void)in_sizes; (void)n_in; (void)out_size; (void)ws_size;

  // workspace carve-up (~34.0 MB)
  char* ws = (char*)d_ws;
  size_t off = 0;
  _Float16* EN_h   = (_Float16*)(ws + off); off += (size_t)BDIM * SDIM * EDIM * 2;
  _Float16* en_h   = (_Float16*)(ws + off); off += (size_t)BDIM * SDIM * EDIM * 2;
  _Float16* Wt_h   = (_Float16*)(ws + off); off += (size_t)EDIM * EDIM * 2;
  float*    L      = (float*)(ws + off);    off += (size_t)BDIM * EDIM * 4;
  float*    SE     = (float*)(ws + off);    off += (size_t)BDIM * NSELFD * EDIM * 4;
  float*    baseP  = (float*)(ws + off);    off += (size_t)BDIM * NSELFD * EDIM * 4;
  _Float16* lstm_h = (_Float16*)(ws + off); off += (size_t)BDIM * EDIM * 2;
  _Float16* se_h   = (_Float16*)(ws + off); off += (size_t)BDIM * NSELFD * EDIM * 2;

  const int MEN = BDIM * SDIM;  // 32768
  dim3 blk(256);

  // stage 1: relu projections (f16 outputs)
  gemm256_wmma<false, true, true><<<dim3(MEN / 128, 4), blk, 0, stream>>>(entity, W_en,   b_en,   en_h,   MEN);
  gemm256_wmma<false, true, true><<<dim3(1, 4),         blk, 0, stream>>>(lstm,   W_lstm, b_lstm, lstm_h, BDIM);
  gemm256_wmma<false, true, true><<<dim3(1, 4),         blk, 0, stream>>>(selfe,  W_self, b_self, se_h,   BDIM * NSELFD);
  // stage 2: W1 projections
  transpose_w_f16<<<dim3(EDIM * EDIM / 256), blk, 0, stream>>>(W1 + (size_t)EDIM * EDIM, Wt_h);
  gemm256_wmma_async<<<dim3(MEN / 128, 4), blk, 0, stream>>>(en_h, Wt_h, EN_h, MEN);
  gemm256_wmma<true, false, false><<<dim3(1, 4), blk, 0, stream>>>(lstm_h, W1,                           nullptr, L,  BDIM);
  gemm256_wmma<true, false, false><<<dim3(1, 4), blk, 0, stream>>>(se_h,   W1 + (size_t)2 * EDIM * EDIM, nullptr, SE, BDIM * NSELFD);
  // stage 3: broadcast bases, then fused relu-GEMM-mask-softmax-argmax
  prep_base<<<dim3(128), blk, 0, stream>>>(L, SE, b1, baseP);

  int*   act  = (int*)d_out;
  float* prob = (float*)d_out + (size_t)BDIM * NSELFD * SDIM;
  attack_fused<<<dim3(BDIM * NSELFD), blk, 0, stream>>>(EN_h, baseP, W2, b2, avail, act, prob);
}
